// PremiseSelectionModel_84215718740066
// MI455X (gfx1250) — matrix operand order repaired and checked by reading, not compiled, not run on hardware
//
#include <hip/hip_runtime.h>

#define DCH     128
#define VOCABSZ 512
#define NLAYERS 3
#define NGRAPH  128
#define BN_EPS  1e-5f
#define LDS_PAD 8     // __bf16 elements of row padding for bank-conflict avoidance

typedef __bf16        v16bf  __attribute__((ext_vector_type(16)));
typedef float         v8f    __attribute__((ext_vector_type(8)));
typedef unsigned int  u32x4  __attribute__((ext_vector_type(4)));
typedef float         f32x4  __attribute__((ext_vector_type(4)));

union BF16Frag { v16bf v; u32x4 q[2]; };

__device__ __forceinline__ __bf16 f2bf(float f) {
  unsigned u = __float_as_uint(f);
  u = u + 0x7FFFu + ((u >> 16) & 1u);          // round-to-nearest-even
  unsigned short h = (unsigned short)(u >> 16);
  return __builtin_bit_cast(__bf16, h);
}

// ---------------------------------------------------------------------------
// argmax over 512-wide one-hot scores -> embedding gather. One wave per node.
// ---------------------------------------------------------------------------
__global__ __launch_bounds__(256)
void argmax_embed_kernel(const float* __restrict__ X, const float* __restrict__ emb,
                         float* __restrict__ xf, __bf16* __restrict__ xb, int N) {
  int lane = threadIdx.x & 31;
  int node = blockIdx.x * 8 + (threadIdx.x >> 5);
  if (node >= N) return;
  const float* row = X + (long)node * VOCABSZ;
  float best = -3.4e38f; int bi = 0;
  for (int k = lane; k < VOCABSZ; k += 32) {
    float v = row[k];
    if (v > best) { best = v; bi = k; }
  }
  for (int off = 16; off > 0; off >>= 1) {
    float ov = __shfl_down(best, off, 32);
    int   oi = __shfl_down(bi,   off, 32);
    if (ov > best || (ov == best && oi < bi)) { best = ov; bi = oi; }
  }
  bi = __shfl(bi, 0, 32);
  const float* er = emb + (long)bi * DCH;
  long base = (long)node * DCH + lane * 4;
  #pragma unroll
  for (int q = 0; q < 4; ++q) {
    float v = er[lane * 4 + q];
    xf[base + q] = v;
    xb[base + q] = f2bf(v);
  }
}

// ---------------------------------------------------------------------------
// Gather + GEMM (bf16 WMMA, f32 accumulate) + BN column statistics.
//   X  : [srcRows x 128] bf16 activations (gathered rows)
//   twi: [3 x twiStride] row indices (K segment selects level), or nullptr
//   Wb : [128 x KIN] bf16, n-major (pre-transposed weights)
// Weight panel staged in LDS once per block (dynamic shared, padded rows).
// Per k-step: next A gather issued, all 8 B fragments loaded into distinct
// registers (sched barriers pin the load group above the WMMA group so the
// allocator cannot collapse them), then 8 WMMAs with partial dscnt waits.
// Block = 256 threads = 8 waves; each wave computes 16 rows x 128 cols.
// ---------------------------------------------------------------------------
template <int KIN>
__global__ __launch_bounds__(256)
void gemm_bn_kernel(const __bf16* __restrict__ X, const int* __restrict__ twi,
                    const __bf16* __restrict__ Wb, float* __restrict__ Hpre,
                    float* __restrict__ colSum, float* __restrict__ colSumSq,
                    int R, int twiStride) {
  constexpr int KP    = KIN + LDS_PAD;         // padded LDS row stride (elems)
  constexpr int NSTEP = KIN / 32;
  extern __shared__ __bf16 smemB[];            // [128 x KP] bf16
  __shared__ float cs[DCH];
  __shared__ float cq[DCH];
  if (threadIdx.x < DCH) { cs[threadIdx.x] = 0.f; cq[threadIdx.x] = 0.f; }

  // cooperative stage: global Wb [128 x KIN] -> LDS [128 x KP]
  constexpr int CHUNKS = (KIN * DCH) / 8;      // 8 bf16 (16B) per chunk
  for (int c = threadIdx.x; c < CHUNKS; c += 256) {
    int n  = c / (KIN / 8);
    int ko = (c % (KIN / 8)) * 8;
    *(u32x4*)(smemB + (size_t)n * KP + ko) = *(const u32x4*)(Wb + (size_t)n * KIN + ko);
  }
  __syncthreads();

  const int lane = threadIdx.x & 31;
  const int wave = threadIdx.x >> 5;
  const int half = lane >> 4;
  const int lnm  = lane & 15;
  const int rowBase = blockIdx.x * 128 + wave * 16;

  int aRow = rowBase + lnm;
  if (aRow > R - 1) aRow = R - 1;
  int src0 = aRow, src1 = aRow, src2 = aRow;
  if (twi) {
    src0 = twi[aRow];
    src1 = twi[twiStride + aRow];
    src2 = twi[2 * twiStride + aRow];
  }

  auto loadA = [&](BF16Frag& f, int k0) {
    int src = src0;
    if (KIN > DCH && twi) {
      int seg = k0 >> 7;                       // compile-time per unrolled iter
      src = (seg == 0) ? src0 : ((seg == 1) ? src1 : src2);
    }
    const __bf16* xp = X + (long)src * DCH + (k0 & (DCH - 1)) + half * 8;
    f.q[0] = *(const u32x4*)(xp);              // K = half*8    + [0..7]
    f.q[1] = *(const u32x4*)(xp + 16);         // K = 16+half*8 + [0..7]
  };

  v8f acc[8];
  #pragma unroll
  for (int t = 0; t < 8; ++t)
    #pragma unroll
    for (int j = 0; j < 8; ++j) acc[t][j] = 0.f;

  BF16Frag a;
  loadA(a, 0);

  #pragma unroll
  for (int kk = 0; kk < NSTEP; ++kk) {
    const int k0 = kk * 32;
    // prefetch next k-step's gathered A fragment under this step's WMMAs
    BF16Frag an;
    if (kk + 1 < NSTEP) loadA(an, k0 + 32);
    // load all 8 B fragments into independent registers
    BF16Frag b[8];
    const __bf16* wp = smemB + (size_t)lnm * KP + k0 + half * 8;
    #pragma unroll
    for (int t = 0; t < 8; ++t) {
      const __bf16* wpt = wp + (size_t)t * 16 * KP;
      b[t].q[0] = *(const u32x4*)(wpt);
      b[t].q[1] = *(const u32x4*)(wpt + 16);
    }
    // pin the load group above the WMMA group: forces 8 live fragments and
    // incremental (partial) dscnt waits instead of a drain per WMMA
    __builtin_amdgcn_sched_barrier(0);
    #pragma unroll
    for (int t = 0; t < 8; ++t) {
      acc[t] = __builtin_amdgcn_wmma_f32_16x16x32_bf16(
          false, a.v, false, b[t].v, (short)0, acc[t], false, false);
    }
    __builtin_amdgcn_sched_barrier(0);
    if (kk + 1 < NSTEP) a = an;
  }

  // Store + per-column partial sums (masked to valid rows)
  #pragma unroll
  for (int t = 0; t < 8; ++t) {
    int col = t * 16 + lnm;
    float s = 0.f, sq = 0.f;
    #pragma unroll
    for (int r = 0; r < 8; ++r) {
      int row = rowBase + half * 8 + r;        // C/D layout: M = r + 8*half
      if (row < R) {
        float v = acc[t][r];
        Hpre[(long)row * DCH + col] = v;
        s += v; sq += v * v;
      }
    }
    atomicAdd(&cs[col], s);
    atomicAdd(&cq[col], sq);
  }
  __syncthreads();
  if (threadIdx.x < DCH) {
    atomicAdd(&colSum[threadIdx.x],   cs[threadIdx.x]);
    atomicAdd(&colSumSq[threadIdx.x], cq[threadIdx.x]);
  }
}

// ---------------------------------------------------------------------------
__global__ void finalize_stats_kernel(const float* __restrict__ colSum,
                                      const float* __restrict__ colSumSq,
                                      const float* __restrict__ gamma,
                                      const float* __restrict__ beta,
                                      float* __restrict__ scale,
                                      float* __restrict__ shift,
                                      float invCount, int C) {
  int c = threadIdx.x;
  if (c < C) {
    float mu  = colSum[c] * invCount;
    float var = colSumSq[c] * invCount - mu * mu;
    float is  = rsqrtf(var + BN_EPS) * gamma[c];
    scale[c] = is;
    shift[c] = beta[c] - mu * is;
  }
}

// BN + relu + scaled scatter-add (reciprocal count folded per edge)
__global__ void bn_scatter_kernel(const float* __restrict__ Hpre,
                                  const float* __restrict__ scale,
                                  const float* __restrict__ shift,
                                  const int* __restrict__ idx,
                                  const float* __restrict__ invCnt,
                                  float* __restrict__ M, int E) {
  long gid = (long)blockIdx.x * blockDim.x + threadIdx.x;
  if (gid >= (long)E * 32) return;
  int e = (int)(gid >> 5);
  int c = (int)(gid & 31) * 4;
  int n = idx[e];
  float w = invCnt[n];
  f32x4 h = *(const f32x4*)(Hpre + (long)e * DCH + c);
  float* mp = M + (long)n * DCH + c;
  atomicAdd(mp + 0, fmaxf(h.x * scale[c + 0] + shift[c + 0], 0.f) * w);
  atomicAdd(mp + 1, fmaxf(h.y * scale[c + 1] + shift[c + 1], 0.f) * w);
  atomicAdd(mp + 2, fmaxf(h.z * scale[c + 2] + shift[c + 2], 0.f) * w);
  atomicAdd(mp + 3, fmaxf(h.w * scale[c + 3] + shift[c + 3], 0.f) * w);
}

// BN + relu + residual add; refresh f32 and bf16 copies of x
__global__ void bn_residual_kernel(const float* __restrict__ h2,
                                   const float* __restrict__ scale,
                                   const float* __restrict__ shift,
                                   float* __restrict__ xf, __bf16* __restrict__ xb,
                                   long nq) {
  long gid = (long)blockIdx.x * blockDim.x + threadIdx.x;
  if (gid >= nq) return;
  long base = gid * 4;
  int c = (int)(base & (DCH - 1));
  f32x4 h = *(const f32x4*)(h2 + base);
  f32x4 x = *(const f32x4*)(xf + base);
  f32x4 r;
  r.x = fmaxf(h.x * scale[c + 0] + shift[c + 0], 0.f) + x.x;
  r.y = fmaxf(h.y * scale[c + 1] + shift[c + 1], 0.f) + x.y;
  r.z = fmaxf(h.z * scale[c + 2] + shift[c + 2], 0.f) + x.z;
  r.w = fmaxf(h.w * scale[c + 3] + shift[c + 3], 0.f) + x.w;
  *(f32x4*)(xf + base) = r;
  xb[base + 0] = f2bf(r.x); xb[base + 1] = f2bf(r.y);
  xb[base + 2] = f2bf(r.z); xb[base + 3] = f2bf(r.w);
}

__global__ void to_bf16_kernel(const float* __restrict__ s, __bf16* __restrict__ d, long n) {
  long i = (long)blockIdx.x * blockDim.x + threadIdx.x;
  if (i < n) d[i] = f2bf(s[i]);
}

__global__ void wtrans_kernel(const float* __restrict__ W, __bf16* __restrict__ Wb, int Kin) {
  int i = blockIdx.x * blockDim.x + threadIdx.x;
  if (i < Kin * DCH) {
    int k = i / DCH, n = i % DCH;
    Wb[(long)n * Kin + k] = f2bf(W[i]);        // [Kin x 128] -> [128 x Kin]
  }
}

__global__ void edge_count_kernel(const int* __restrict__ twi, float* __restrict__ cnt,
                                  int N, int E) {
  int e = blockIdx.x * blockDim.x + threadIdx.x;
  if (e < E) {
    atomicAdd(&cnt[twi[e]], 1.f);
    atomicAdd(&cnt[N + twi[E + e]], 1.f);
    atomicAdd(&cnt[2 * N + twi[2 * E + e]], 1.f);
  }
}

__global__ void batch_count_kernel(const int* __restrict__ batch, float* __restrict__ bc, int N) {
  int n = blockIdx.x * blockDim.x + threadIdx.x;
  if (n < N) atomicAdd(&bc[batch[n]], 1.f);
}

__global__ void invert_kernel(float* __restrict__ c, long n) {
  long i = (long)blockIdx.x * blockDim.x + threadIdx.x;
  if (i < n) c[i] = 1.f / fmaxf(c[i], 1.f);
}

__global__ void pool_kernel(const float* __restrict__ xf, const int* __restrict__ batch,
                            const float* __restrict__ invB, float* __restrict__ g, int N) {
  long gid = (long)blockIdx.x * blockDim.x + threadIdx.x;
  if (gid >= (long)N * 32) return;
  int n = (int)(gid >> 5);
  int c = (int)(gid & 31) * 4;
  int b = batch[n];
  float w = invB[b];
  f32x4 x = *(const f32x4*)(xf + (long)n * DCH + c);
  float* gp = g + (long)b * DCH + c;
  atomicAdd(gp + 0, x.x * w);
  atomicAdd(gp + 1, x.y * w);
  atomicAdd(gp + 2, x.z * w);
  atomicAdd(gp + 3, x.w * w);
}

// ---------------------------- tiny classifier ------------------------------
__global__ void cls_gemm1_kernel(const float* __restrict__ gs, const float* __restrict__ gt,
                                 const float* __restrict__ W1, float* __restrict__ h1,
                                 float* __restrict__ colSum, float* __restrict__ colSumSq) {
  int r = blockIdx.x, c = threadIdx.x;
  float acc = 0.f;
  for (int k = 0; k < DCH; ++k) acc += gs[r * DCH + k] * W1[k * DCH + c];
  for (int k = 0; k < DCH; ++k) acc += gt[r * DCH + k] * W1[(DCH + k) * DCH + c];
  h1[r * DCH + c] = acc;
  atomicAdd(&colSum[c], acc);
  atomicAdd(&colSumSq[c], acc * acc);
}

__global__ void cls_gemm2_kernel(const float* __restrict__ h1, const float* __restrict__ scale,
                                 const float* __restrict__ shift, const float* __restrict__ W2,
                                 float* __restrict__ outpre, float* __restrict__ colSum,
                                 float* __restrict__ colSumSq) {
  __shared__ float z[DCH];
  int r = blockIdx.x, t = threadIdx.x;
  z[t] = fmaxf(h1[r * DCH + t] * scale[t] + shift[t], 0.f);
  __syncthreads();
  if (t < 2) {
    float acc = 0.f;
    for (int k = 0; k < DCH; ++k) acc += z[k] * W2[k * 2 + t];
    outpre[r * 2 + t] = acc;
    atomicAdd(&colSum[t], acc);
    atomicAdd(&colSumSq[t], acc * acc);
  }
}

__global__ void cls_out_kernel(const float* __restrict__ outpre, const float* __restrict__ scale,
                               const float* __restrict__ shift, float* __restrict__ out) {
  int i = threadIdx.x;
  if (i < NGRAPH * 2) {
    int j = i & 1;
    out[i] = outpre[i] * scale[j] + shift[j];
  }
}

// ---------------------------------------------------------------------------
extern "C" void kernel_launch(void* const* d_in, const int* in_sizes, int n_in,
                              void* d_out, int out_size, void* d_ws, size_t ws_size,
                              hipStream_t stream) {
  const int N = in_sizes[0] / VOCABSZ;
  const int E = in_sizes[27] / 3;

  const float* emb = (const float*)d_in[2];

  // workspace carve-out
  size_t off = 0;
  auto alloc = [&](size_t bytes) -> void* {
    void* p = (char*)d_ws + off;
    off += (bytes + 255) & ~(size_t)255;
    return p;
  };
  __bf16* xb     = (__bf16*)alloc((size_t)N * DCH * 2);
  float*  xf     = (float*) alloc((size_t)N * DCH * 4);
  __bf16* WbBig  = (__bf16*)alloc((size_t)9 * DCH * 384 * 2);
  __bf16* WbTW   = (__bf16*)alloc((size_t)NLAYERS * DCH * DCH * 2);
  float*  hpre   = (float*) alloc((size_t)E * DCH * 4);
  float*  mbuf   = (float*) alloc((size_t)N * DCH * 4);
  __bf16* mbf    = (__bf16*)alloc((size_t)N * DCH * 2);
  float*  h2     = (float*) alloc((size_t)N * DCH * 4);
  float*  cnt    = (float*) alloc((size_t)3 * N * 4);
  float*  bcnt   = (float*) alloc((size_t)NGRAPH * 4);
  float*  colSum   = (float*)alloc(DCH * 4);
  float*  colSumSq = (float*)alloc(DCH * 4);
  float*  scale  = (float*) alloc(DCH * 4);
  float*  shift  = (float*) alloc(DCH * 4);
  float*  g_s    = (float*) alloc((size_t)NGRAPH * DCH * 4);
  float*  g_t    = (float*) alloc((size_t)NGRAPH * DCH * 4);
  float*  h1     = (float*) alloc((size_t)NGRAPH * DCH * 4);
  float*  outpre = (float*) alloc((size_t)NGRAPH * 2 * 4);

  const size_t ldsBig = (size_t)DCH * (384 + LDS_PAD) * sizeof(__bf16); // ~98 KB
  const size_t ldsTW  = (size_t)DCH * (DCH + LDS_PAD) * sizeof(__bf16); // ~34 KB

  // ---- pre-transpose weights to bf16 [n][k] layout ----
  for (int i = 0; i < NLAYERS; ++i) {
    for (int m = 0; m < 3; ++m) {              // WT=3, WM=7, WB=11
      const float* W = (const float*)d_in[3 + 4 * m] + (size_t)i * 384 * DCH;
      wtrans_kernel<<<(384 * DCH + 255) / 256, 256, 0, stream>>>(
          W, WbBig + (size_t)(i * 3 + m) * DCH * 384, 384);
    }
    const float* Wtw = (const float*)d_in[15] + (size_t)i * DCH * DCH;
    wtrans_kernel<<<(DCH * DCH + 255) / 256, 256, 0, stream>>>(
        Wtw, WbTW + (size_t)i * DCH * DCH, DCH);
  }

  // ---- two sides (shared DAG parameters) ----
  for (int side = 0; side < 2; ++side) {
    const float* xoh   = (const float*)d_in[side];
    const int*   twi   = (const int*)  d_in[27 + side];
    const int*   batch = (const int*)  d_in[29 + side];
    float* gpool = side ? g_t : g_s;

    argmax_embed_kernel<<<(N + 7) / 8, 256, 0, stream>>>(xoh, emb, xf, xb, N);

    hipMemsetAsync(cnt, 0, (size_t)3 * N * 4, stream);
    edge_count_kernel<<<(E + 255) / 256, 256, 0, stream>>>(twi, cnt, N, E);
    invert_kernel<<<(3 * N + 255) / 256, 256, 0, stream>>>(cnt, (long)3 * N);

    for (int i = 0; i < NLAYERS; ++i) {
      hipMemsetAsync(mbuf, 0, (size_t)N * DCH * 4, stream);
      for (int m = 0; m < 3; ++m) {
        hipMemsetAsync(colSum,   0, DCH * 4, stream);
        hipMemsetAsync(colSumSq, 0, DCH * 4, stream);
        gemm_bn_kernel<384><<<(E + 127) / 128, 256, ldsBig, stream>>>(
            xb, twi, WbBig + (size_t)(i * 3 + m) * DCH * 384,
            hpre, colSum, colSumSq, E, E);
        const float* gamma = (const float*)d_in[5 + 4 * m] + i * DCH;
        const float* beta  = (const float*)d_in[6 + 4 * m] + i * DCH;
        finalize_stats_kernel<<<1, DCH, 0, stream>>>(
            colSum, colSumSq, gamma, beta, scale, shift, 1.0f / (float)E, DCH);
        bn_scatter_kernel<<<(int)(((long)E * 32 + 255) / 256), 256, 0, stream>>>(
            hpre, scale, shift, twi + (size_t)m * E, cnt + (size_t)m * N, mbuf, E);
      }
      to_bf16_kernel<<<(int)(((long)N * DCH + 255) / 256), 256, 0, stream>>>(
          mbuf, mbf, (long)N * DCH);
      hipMemsetAsync(colSum,   0, DCH * 4, stream);
      hipMemsetAsync(colSumSq, 0, DCH * 4, stream);
      gemm_bn_kernel<DCH><<<(N + 127) / 128, 256, ldsTW, stream>>>(
          mbf, (const int*)nullptr, WbTW + (size_t)i * DCH * DCH,
          h2, colSum, colSumSq, N, 0);
      finalize_stats_kernel<<<1, DCH, 0, stream>>>(
          colSum, colSumSq, (const float*)d_in[17] + i * DCH,
          (const float*)d_in[18] + i * DCH, scale, shift, 1.0f / (float)N, DCH);
      bn_residual_kernel<<<(int)(((long)N * 32 + 255) / 256), 256, 0, stream>>>(
          h2, scale, shift, xf, xb, (long)N * 32);
    }

    hipMemsetAsync(bcnt, 0, NGRAPH * 4, stream);
    batch_count_kernel<<<(N + 255) / 256, 256, 0, stream>>>(batch, bcnt, N);
    invert_kernel<<<1, 256, 0, stream>>>(bcnt, NGRAPH);
    hipMemsetAsync(gpool, 0, (size_t)NGRAPH * DCH * 4, stream);
    pool_kernel<<<(int)(((long)N * 32 + 255) / 256), 256, 0, stream>>>(
        xf, batch, bcnt, gpool, N);
  }

  // ---- classifier ----
  hipMemsetAsync(colSum,   0, DCH * 4, stream);
  hipMemsetAsync(colSumSq, 0, DCH * 4, stream);
  cls_gemm1_kernel<<<NGRAPH, DCH, 0, stream>>>(
      g_s, g_t, (const float*)d_in[19], h1, colSum, colSumSq);
  finalize_stats_kernel<<<1, DCH, 0, stream>>>(
      colSum, colSumSq, (const float*)d_in[21], (const float*)d_in[22],
      scale, shift, 1.0f / (float)NGRAPH, DCH);
  hipMemsetAsync(colSum,   0, DCH * 4, stream);
  hipMemsetAsync(colSumSq, 0, DCH * 4, stream);
  cls_gemm2_kernel<<<NGRAPH, DCH, 0, stream>>>(
      h1, scale, shift, (const float*)d_in[23], outpre, colSum, colSumSq);
  finalize_stats_kernel<<<1, 2, 0, stream>>>(
      colSum, colSumSq, (const float*)d_in[25], (const float*)d_in[26],
      scale, shift, 1.0f / (float)NGRAPH, 2);
  cls_out_kernel<<<1, 256, 0, stream>>>(outpre, scale, shift, (float*)d_out);
}
